// CriteoMLP_37477884625195
// MI455X (gfx1250) — compile-verified, matrix operand order checked by reading, not run on hardware
//
#include <hip/hip_runtime.h>
#include <hip/hip_bf16.h>
#include <stdint.h>

typedef _Float16 h16;
typedef __attribute__((ext_vector_type(4)))  _Float16 v4h;
typedef __attribute__((ext_vector_type(8)))  _Float16 v8h;
typedef __attribute__((ext_vector_type(16))) _Float16 v16h;
typedef __attribute__((ext_vector_type(8)))  float    v8f;
typedef __attribute__((ext_vector_type(4)))  int      v4i;

#define BM 128
#define BN 64
#define SCS 72  // C-staging LDS row stride in halves (64 + 8 pad)

#if defined(__gfx1250__) && __has_builtin(__builtin_amdgcn_global_load_async_to_lds_b128)
#define USE_ASYNC 1
#endif

#ifdef USE_ASYNC
typedef __attribute__((address_space(1))) v4i gv4i;  // global <4 x i32>
typedef __attribute__((address_space(3))) v4i lv4i;  // LDS    <4 x i32>
#endif

__device__ __forceinline__ void cp16(const void* g, void* l) {
#ifdef USE_ASYNC
  __builtin_amdgcn_global_load_async_to_lds_b128((gv4i*)(g), (lv4i*)(l), 0, 0);
#else
  *(v8h*)l = *(const v8h*)g;
#endif
}

__device__ __forceinline__ void async_wait0() {
#ifdef USE_ASYNC
#if __has_builtin(__builtin_amdgcn_s_wait_asynccnt)
  __builtin_amdgcn_s_wait_asynccnt(0);
#else
  asm volatile("s_wait_asynccnt 0" ::: "memory");
#endif
#endif
}

__device__ __forceinline__ v16h mk16(v8h lo, v8h hi) {
  return __builtin_shufflevector(lo, hi, 0,1,2,3,4,5,6,7,8,9,10,11,12,13,14,15);
}

__device__ __forceinline__ v8f wmma16(v16h a, v16h b, v8f c) {
#if defined(__gfx1250__)
  // D(f32) = A(16x32 f16) * B(32x16 f16) + C
  return __builtin_amdgcn_wmma_f32_16x16x32_f16(false, a, false, b, (short)0, c,
                                                false, false);
#else
  return c;  // host-compilation stub
#endif
}

// ---------------------------------------------------------------------------
// Fused GEMM tile kernel.  Block tile 128(M) x 64(N), 256 threads = 8 waves
// arranged 4(M) x 2(N); each wave owns a 32x32 output = 2x2 WMMA tiles.
// EMBED=true: A built in LDS from the fused embedding gather (K=288, padded).
// EMBED=false: A async-copied from the previous layer's f16 activations.
// Epilogue: +bias, ReLU, f16 store (LDS-staged, coalesced b128), and
// per-column sum/sumsq via ds_add_f32 then one global atomic per column.
// ---------------------------------------------------------------------------
template <bool EMBED>
__global__ __launch_bounds__(256)
void k_gemm(const h16* __restrict__ A, const int* __restrict__ x,
            const float* __restrict__ tables, const h16* __restrict__ Bt,
            const float* __restrict__ bias, h16* __restrict__ C,
            float* __restrict__ gsum, float* __restrict__ gsq, int N, int K) {
  extern __shared__ char smem[];
  const int KA = K + 8;  // half-element row stride: keeps 16B alignment, skews banks
  h16* sA = (h16*)smem;              // [BM][KA]
  h16* sB = sA + BM * KA;            // [BN][KA]  (weights pre-transposed: [n][k])
  float* sSum = (float*)(sB + BN * KA);
  float* sSq = sSum + BN;
  h16* sC = sA;                      // reuse A region for C staging [BM][SCS]

  const int tid = threadIdx.x;
  const int row0 = blockIdx.x * BM;
  const int col0 = blockIdx.y * BN;

  if (tid < BN) { sSum[tid] = 0.f; sSq[tid] = 0.f; }

  if constexpr (EMBED) {
    // 128 rows x 18 groups of 16 halves (group 17 = zero K-padding 272..287)
    for (int i = tid; i < BM * 18; i += 256) {
      int r = i / 18, t = i - r * 18;
      h16* dst = &sA[r * KA + t * 16];
      if (t < 17) {
        int idx = x[(size_t)(row0 + r) * 17 + t];
        const float4* s4 = (const float4*)(tables + ((size_t)t * 512 + idx) * 16);
        float4 f0 = s4[0], f1 = s4[1], f2 = s4[2], f3 = s4[3];
        v8h h0, h1;
        h0[0] = (h16)f0.x; h0[1] = (h16)f0.y; h0[2] = (h16)f0.z; h0[3] = (h16)f0.w;
        h0[4] = (h16)f1.x; h0[5] = (h16)f1.y; h0[6] = (h16)f1.z; h0[7] = (h16)f1.w;
        h1[0] = (h16)f2.x; h1[1] = (h16)f2.y; h1[2] = (h16)f2.z; h1[3] = (h16)f2.w;
        h1[4] = (h16)f3.x; h1[5] = (h16)f3.y; h1[6] = (h16)f3.z; h1[7] = (h16)f3.w;
        ((v8h*)dst)[0] = h0; ((v8h*)dst)[1] = h1;
      } else {
        v8h z = {};
        ((v8h*)dst)[0] = z; ((v8h*)dst)[1] = z;
      }
    }
  } else {
    const int cpr = K >> 3;  // 16B chunks per row
    for (int i = tid; i < BM * cpr; i += 256) {
      int r = i / cpr, c = i - r * cpr;
      cp16(&A[(size_t)(row0 + r) * K + c * 8], &sA[r * KA + c * 8]);
    }
  }
  {
    const int cpr = K >> 3;
    for (int i = tid; i < BN * cpr; i += 256) {
      int r = i / cpr, c = i - r * cpr;
      cp16(&Bt[(size_t)(col0 + r) * K + c * 8], &sB[r * KA + c * 8]);
    }
  }
  async_wait0();
  __syncthreads();

  const int lane = tid & 31;
  const int wave = tid >> 5;
  const int wm = wave >> 1, wn = wave & 1;
  const int lr = lane & 15;
  const int khi = (lane & 16) ? 8 : 0;  // K-half split per WMMA 16-bit layout

  const h16* a0p = &sA[(wm * 32 + lr) * KA];
  const h16* a1p = a0p + 16 * KA;
  const h16* b0p = &sB[(wn * 32 + lr) * KA];
  const h16* b1p = b0p + 16 * KA;

  v8f z8 = {0.f, 0.f, 0.f, 0.f, 0.f, 0.f, 0.f, 0.f};
  v8f acc00 = z8, acc01 = z8, acc10 = z8, acc11 = z8;

  for (int kk = 0; kk < K; kk += 32) {
    const int o0 = kk + khi, o1 = o0 + 16;
    v16h va0 = mk16(*(const v8h*)(a0p + o0), *(const v8h*)(a0p + o1));
    v16h va1 = mk16(*(const v8h*)(a1p + o0), *(const v8h*)(a1p + o1));
    v16h vb0 = mk16(*(const v8h*)(b0p + o0), *(const v8h*)(b0p + o1));
    v16h vb1 = mk16(*(const v8h*)(b1p + o0), *(const v8h*)(b1p + o1));
    acc00 = wmma16(va0, vb0, acc00);
    acc01 = wmma16(va0, vb1, acc01);
    acc10 = wmma16(va1, vb0, acc10);
    acc11 = wmma16(va1, vb1, acc11);
  }
  __syncthreads();  // everyone done reading sA/sB before sC overwrites

  auto epi = [&](v8f c, int mi, int ni) {
    int col = wn * 32 + ni * 16 + lr;
    float bv = bias[col0 + col];
    int rb = wm * 32 + mi * 16 + ((lane & 16) ? 8 : 0);
    float s1 = 0.f, s2 = 0.f;
#pragma unroll
    for (int r = 0; r < 8; ++r) {
      float v = c[r] + bv;
      v = v > 0.f ? v : 0.f;
      s1 += v; s2 += v * v;
      sC[(rb + r) * SCS + col] = (h16)v;
    }
    atomicAdd(&sSum[col], s1);
    atomicAdd(&sSq[col], s2);
  };
  epi(acc00, 0, 0); epi(acc01, 0, 1); epi(acc10, 1, 0); epi(acc11, 1, 1);
  __syncthreads();

  {  // coalesced f16 store: 2 threads per row, 32 halves (4 x b128) each
    int r = tid >> 1, cb = (tid & 1) * 32;
    const v8h* s = (const v8h*)&sC[r * SCS + cb];
    v8h* d = (v8h*)&C[(size_t)(row0 + r) * N + col0 + cb];
    d[0] = s[0]; d[1] = s[1]; d[2] = s[2]; d[3] = s[3];
  }
  if (tid < BN) {
    atomicAdd(&gsum[col0 + tid], sSum[tid]);
    atomicAdd(&gsq[col0 + tid], sSq[tid]);
  }
}

// sums/sumsq -> per-feature BN scale/shift
__global__ void k_bn(const float* __restrict__ sums, const float* __restrict__ sqs,
                     const float* __restrict__ g, const float* __restrict__ be,
                     float* __restrict__ scale, float* __restrict__ shift,
                     float invB, int N) {
  int n = threadIdx.x;
  if (n < N) {
    float mu = sums[n] * invB;
    float var = sqs[n] * invB - mu * mu;
    float rs = rsqrtf(var + 1e-5f);
    float sc = g[n] * rs;
    scale[n] = sc;
    shift[n] = be[n] - mu * sc;
  }
}

// Fold BN(prev layer) into next layer's weights:
//   Wt[n][k] = (f16)(scale[k] * W[k][n]);  bf[n] = b[n] + sum_k shift[k]*W[k][n]
// launch: grid = N blocks, block = K threads (K in {128,256})
__global__ void k_fold(const float* __restrict__ W, const float* __restrict__ b,
                       const float* __restrict__ scale, const float* __restrict__ shift,
                       h16* __restrict__ Wt, float* __restrict__ bf, int K, int N) {
  __shared__ float red[256];
  int n = blockIdx.x, k = threadIdx.x;
  float w = W[(size_t)k * N + n];
  Wt[(size_t)n * K + k] = (h16)(scale[k] * w);
  red[k] = shift[k] * w;
  __syncthreads();
  for (int off = blockDim.x >> 1; off > 0; off >>= 1) {
    if (k < off) red[k] += red[k + off];
    __syncthreads();
  }
  if (k == 0) bf[n] = b[n] + red[0];
}

// W0 f32 [272][256] -> transposed, K-padded f16 [256][288]
__global__ void k_convw0(const float* __restrict__ W, h16* __restrict__ Wt) {
  int id = blockIdx.x * 256 + threadIdx.x;
  if (id >= 256 * 288) return;
  int n = id / 288, k = id - n * 288;
  Wt[id] = (k < 272) ? (h16)W[(size_t)k * 256 + n] : (h16)0.f;
}

__global__ void k_zero(float* __restrict__ p, int n) {
  int i = blockIdx.x * 256 + threadIdx.x;
  if (i < n) p[i] = 0.f;
}

// head: one wave per row, lane covers 4 of the 128 features
__global__ __launch_bounds__(256)
void k_out(const h16* __restrict__ A, const h16* __restrict__ wf,
           const float* __restrict__ c0, float* __restrict__ out, int M) {
  __shared__ float swf[128];
  if (threadIdx.x < 128) swf[threadIdx.x] = (float)wf[threadIdx.x];
  __syncthreads();
  const int lane = threadIdx.x & 31;
  const int wave = threadIdx.x >> 5;
  const float cc = *c0;
  const int nw = gridDim.x * 8;
  for (int row = blockIdx.x * 8 + wave; row < M; row += nw) {
    v4h av = *(const v4h*)(A + (size_t)row * 128 + lane * 4);
    int kb = lane * 4;
    float d = (float)av[0] * swf[kb] + (float)av[1] * swf[kb + 1] +
              (float)av[2] * swf[kb + 2] + (float)av[3] * swf[kb + 3];
    for (int off = 16; off > 0; off >>= 1) d += __shfl_xor(d, off, 32);
    if (lane == 0) out[row] = d + cc;
  }
}

extern "C" void kernel_launch(void* const* d_in, const int* in_sizes, int n_in,
                              void* d_out, int out_size, void* d_ws, size_t ws_size,
                              hipStream_t stream) {
  const int* x = (const int*)d_in[0];
  const float* tables = (const float*)d_in[1];
  const float* W0 = (const float*)d_in[2];
  const float* b0 = (const float*)d_in[3];
  const float* g0 = (const float*)d_in[4];
  const float* be0 = (const float*)d_in[5];
  const float* W1 = (const float*)d_in[6];
  const float* b1 = (const float*)d_in[7];
  const float* g1 = (const float*)d_in[8];
  const float* be1 = (const float*)d_in[9];
  const float* W2 = (const float*)d_in[10];
  const float* b2 = (const float*)d_in[11];
  const float* g2 = (const float*)d_in[12];
  const float* be2 = (const float*)d_in[13];
  const float* Wo = (const float*)d_in[14];
  const float* bo = (const float*)d_in[15];
  float* out = (float*)d_out;
  const int B = in_sizes[0] / 17;

  char* p = (char*)d_ws;
  auto carve = [&](size_t bytes) -> void* {
    void* r = (void*)p;
    p += (bytes + 255) & ~(size_t)255;
    return r;
  };
  h16* a1 = (h16*)carve((size_t)B * 256 * 2);
  h16* a2 = (h16*)carve((size_t)B * 256 * 2);
  h16* a3 = (h16*)carve((size_t)B * 128 * 2);
  h16* Wt0 = (h16*)carve(256 * 288 * 2);
  h16* Wt1 = (h16*)carve(256 * 256 * 2);
  h16* Wt2 = (h16*)carve(128 * 256 * 2);
  h16* Wt3 = (h16*)carve(128 * 2);
  float* b1f = (float*)carve(256 * 4);
  float* b2f = (float*)carve(128 * 4);
  float* b3f = (float*)carve(4);
  float* stats = (float*)carve(1280 * 4);
  float *sums1 = stats, *sq1 = stats + 256, *sums2 = stats + 512,
        *sq2 = stats + 768, *sums3 = stats + 1024, *sq3 = stats + 1152;
  float* sc1 = (float*)carve(256 * 4); float* sh1 = (float*)carve(256 * 4);
  float* sc2 = (float*)carve(256 * 4); float* sh2 = (float*)carve(256 * 4);
  float* sc3 = (float*)carve(128 * 4); float* sh3 = (float*)carve(128 * 4);

  const float invB = 1.0f / (float)B;
  const size_t smem_e = (size_t)(BM + BN) * (288 + 8) * 2 + BN * 2 * 4;  // ~112 KB
  const size_t smem_m = (size_t)(BM + BN) * (256 + 8) * 2 + BN * 2 * 4;  // ~100 KB

  k_zero<<<dim3((1280 + 255) / 256), dim3(256), 0, stream>>>(stats, 1280);
  k_convw0<<<dim3((256 * 288 + 255) / 256), dim3(256), 0, stream>>>(W0, Wt0);

  // layer 0: gather + GEMM(272->256) + relu, stats
  k_gemm<true><<<dim3(B / BM, 256 / BN), dim3(256), smem_e, stream>>>(
      nullptr, x, tables, Wt0, b0, a1, sums1, sq1, 256, 288);
  k_bn<<<dim3(1), dim3(256), 0, stream>>>(sums1, sq1, g0, be0, sc1, sh1, invB, 256);
  k_fold<<<dim3(256), dim3(256), 0, stream>>>(W1, b1, sc1, sh1, Wt1, b1f, 256, 256);

  // layer 1: GEMM(256->256) with BN folded into weights
  k_gemm<false><<<dim3(B / BM, 256 / BN), dim3(256), smem_m, stream>>>(
      a1, nullptr, nullptr, Wt1, b1f, a2, sums2, sq2, 256, 256);
  k_bn<<<dim3(1), dim3(256), 0, stream>>>(sums2, sq2, g1, be1, sc2, sh2, invB, 256);
  k_fold<<<dim3(128), dim3(256), 0, stream>>>(W2, b2, sc2, sh2, Wt2, b2f, 256, 128);

  // layer 2: GEMM(256->128)
  k_gemm<false><<<dim3(B / BM, 128 / BN), dim3(256), smem_m, stream>>>(
      a2, nullptr, nullptr, Wt2, b2f, a3, sums3, sq3, 128, 256);
  k_bn<<<dim3(1), dim3(128), 0, stream>>>(sums3, sq3, g2, be2, sc3, sh3, invB, 128);
  k_fold<<<dim3(1), dim3(128), 0, stream>>>(Wo, bo, sc3, sh3, Wt3, b3f, 128, 1);

  // head: BN-folded 128->1 dot per row
  k_out<<<dim3(2048), dim3(256), 0, stream>>>(a3, Wt3, b3f, out, B);

  (void)n_in; (void)out_size; (void)ws_size;
}
